// GRRN_15839839387963
// MI455X (gfx1250) — matrix operand order repaired and checked by reading, not compiled, not run on hardware
//
#include <hip/hip_runtime.h>
#include <math.h>

// ---------------------------------------------------------------------------
// Types for CDNA5 WMMA (wave32): 16x16x32 bf16 -> f32
// ---------------------------------------------------------------------------
typedef __attribute__((ext_vector_type(16))) __bf16 v16bf;
typedef __attribute__((ext_vector_type(8)))  float  v8f;

#define TILE_M 128
#define TILE_N 128
#define TILE_K 32
#define LDS_STRIDE 36   // 32 + pad; 72B row pitch -> 4B aligned, conflict-free

// ---------------------------------------------------------------------------
// Generic GEMM with batching (strideB/strideC) and split-K (kChunk):
//   C[z] = act( A[:, z*kChunk:+K] @ B[z](KxN) + bias[z] )
// f32 operands converted to bf16 while staging into LDS; f32 accumulate.
// Double-buffered LDS, pipeline unrolled x2 with compile-time buffers and a
// branch-free steady-state body (tail peeled) so accumulators stay in place
// across the back-edge (no register-reconciliation movs).
// Requires K % 64 == 0 and N % 128 == 0 (true at all call sites).
// ---------------------------------------------------------------------------
__global__ __launch_bounds__(256)
void gemm_bf16_wmma(const float* __restrict__ A, int lda, int kChunk,
                    const float* __restrict__ B, int ldb, long long strideB,
                    const float* __restrict__ bias, long long strideBias,
                    float* __restrict__ C, int ldc, long long strideC,
                    int M, int N, int K, int act)
{
    __shared__ alignas(16) __bf16 AsBuf[2][TILE_M * LDS_STRIDE];  // [m][k]
    __shared__ alignas(16) __bf16 BsBuf[2][TILE_N * LDS_STRIDE];  // [n][k]

    const int t    = threadIdx.x;
    const int lane = t & 31;
    const int wave = t >> 5;
    const int l15  = lane & 15;
    const int lh   = lane >> 4;
    const int mBase = blockIdx.y * TILE_M;
    const int nBase = blockIdx.x * TILE_N;
    const int Mm1   = M - 1;

    A += (long long)blockIdx.z * kChunk;
    B += (long long)blockIdx.z * ((long long)kChunk * ldb + strideB);
    C += (long long)blockIdx.z * strideC;
    const float* biasz = bias ? (bias + (long long)blockIdx.z * strideBias) : nullptr;

    const int waveM = (wave >> 2) * 64;   // 2 waves along M
    const int waveN = (wave & 3) * 32;    // 4 waves along N

    v8f acc[4][2];
    v8f vzero = {};
    for (int i = 0; i < 4; ++i)
        for (int j = 0; j < 2; ++j) acc[i][j] = vzero;

    float4 ra[4];        // staged A tile (128x32 f32), 16 floats/thread
    float4 rb[2][2];     // staged B tile (32x128 f32), two k-rows per slot

    // ---- per-thread running global pointers (advance by one K-tile per call)
    const float* pA[4];
    const float* pBe[2];
    const float* pBo[2];
    #pragma unroll
    for (int i = 0; i < 4; ++i) {
        int linear = t + i * 256;            // 0..1023
        int row = linear >> 3;               // 8 float4 per row
        int cc  = (linear & 7) << 2;
        int gr  = mBase + row;
        gr = gr < Mm1 ? gr : Mm1;            // clamp: OOB rows discarded later
        pA[i] = A + (long long)gr * lda + cc;
    }
    #pragma unroll
    for (int i = 0; i < 2; ++i) {
        int linear = t + i * 256;            // 0..511
        int kp = linear >> 5;                // k-pair index (0..15)
        int nn = (linear & 31) << 2;
        pBe[i] = B + (long long)(2 * kp) * ldb + nBase + nn;
        pBo[i] = pBe[i] + ldb;
    }
    const long long bStep = (long long)TILE_K * ldb;

    auto stage_global = [&]() {
        #pragma unroll
        for (int i = 0; i < 4; ++i) { ra[i] = *(const float4*)pA[i]; pA[i] += TILE_K; }
        #pragma unroll
        for (int i = 0; i < 2; ++i) {
            rb[i][0] = *(const float4*)pBe[i]; pBe[i] += bStep;
            rb[i][1] = *(const float4*)pBo[i]; pBo[i] += bStep;
        }
    };

    // ---- convert staged registers to bf16 and store into LDS buffer
    union PackBF { __bf16 h[2]; unsigned int u; };
    auto stage_lds = [&](__bf16* As, __bf16* Bs) {
        #pragma unroll
        for (int i = 0; i < 4; ++i) {
            int linear = t + i * 256;
            int row = linear >> 3;
            int cc  = (linear & 7) << 2;
            __bf16* dst = &As[row * LDS_STRIDE + cc];
            dst[0] = (__bf16)ra[i].x; dst[1] = (__bf16)ra[i].y;
            dst[2] = (__bf16)ra[i].z; dst[3] = (__bf16)ra[i].w;
        }
        #pragma unroll
        for (int i = 0; i < 2; ++i) {
            int linear = t + i * 256;
            int kp = linear >> 5;
            int nn = (linear & 31) << 2;
            const float* f0 = &rb[i][0].x;
            const float* f1 = &rb[i][1].x;
            #pragma unroll
            for (int j = 0; j < 4; ++j) {
                PackBF p;
                p.h[0] = (__bf16)f0[j];
                p.h[1] = (__bf16)f1[j];
                *(unsigned int*)(&Bs[(nn + j) * LDS_STRIDE + 2 * kp]) = p.u;
            }
        }
    };

    // ---- fragment loads (documented 16-bit VGPR layouts) + 8 WMMAs
    auto compute = [&](const __bf16* As, const __bf16* Bs) {
        v16bf afr[4], bfr[2];
        #pragma unroll
        for (int tm = 0; tm < 4; ++tm) {
            const __bf16* base = &As[(waveM + tm * 16 + l15) * LDS_STRIDE];
            v16bf a;
            #pragma unroll
            for (int vv = 0; vv < 8; ++vv) {
                // A 16x32: VGPR vv holds K = g*16 + h*8 + j*2 (+1)
                int kk = ((vv >> 2) << 4) + (lh << 3) + ((vv & 3) << 1);
                a[2 * vv]     = base[kk];
                a[2 * vv + 1] = base[kk + 1];
            }
            afr[tm] = a;
        }
        #pragma unroll
        for (int tn = 0; tn < 2; ++tn) {
            const __bf16* base = &Bs[(waveN + tn * 16 + l15) * LDS_STRIDE];
            v16bf b;
            #pragma unroll
            for (int vv = 0; vv < 8; ++vv) {
                // B 32x16: VGPR vv holds K = g*16 + h*8 + j*2 (+1)
                int kk = ((vv >> 2) << 4) + (lh << 3) + ((vv & 3) << 1);
                b[2 * vv]     = base[kk];
                b[2 * vv + 1] = base[kk + 1];
            }
            bfr[tn] = b;
        }
        #pragma unroll
        for (int tm = 0; tm < 4; ++tm)
            #pragma unroll
            for (int tn = 0; tn < 2; ++tn)
                acc[tm][tn] = __builtin_amdgcn_wmma_f32_16x16x32_bf16(
                    false, afr[tm], false, bfr[tn], (short)0, acc[tm][tn],
                    false, false);
    };

    // ---- software pipeline: prologue, branch-free steady state, peeled tail
    const int nIter = K / TILE_K;            // even (K % 64 == 0)
    stage_global();                          // tile 0
    stage_lds(AsBuf[0], BsBuf[0]);
    __syncthreads();
    for (int it = 0; it < nIter - 2; it += 2) {
        stage_global();                      // tile it+1
        compute(AsBuf[0], BsBuf[0]);         // tile it
        stage_lds(AsBuf[1], BsBuf[1]);
        __syncthreads();
        stage_global();                      // tile it+2
        compute(AsBuf[1], BsBuf[1]);         // tile it+1
        stage_lds(AsBuf[0], BsBuf[0]);
        __syncthreads();
    }
    // tail: buf0 holds tile nIter-2; tile nIter-1 not yet staged
    stage_global();                          // tile nIter-1
    compute(AsBuf[0], BsBuf[0]);             // tile nIter-2
    stage_lds(AsBuf[1], BsBuf[1]);
    __syncthreads();
    compute(AsBuf[1], BsBuf[1]);             // tile nIter-1

    // ---- epilogue: bias + activation + store (C layout: M = vv + 8*lh, N = l15)
    #pragma unroll
    for (int tm = 0; tm < 4; ++tm) {
        #pragma unroll
        for (int tn = 0; tn < 2; ++tn) {
            int col = nBase + waveN + tn * 16 + l15;
            float bv = biasz ? biasz[col] : 0.f;
            #pragma unroll
            for (int vv = 0; vv < 8; ++vv) {
                int row = mBase + waveM + tm * 16 + vv + 8 * lh;
                if (row < M) {
                    float val = acc[tm][tn][vv] + bv;
                    if (act == 1)      val = 1.f / (1.f + __expf(-val));
                    else if (act == 2) val = tanhf(val);
                    C[(long long)row * ldc + col] = val;
                }
            }
        }
    }
}

// ---------------------------------------------------------------------------
// Split-K reduction: out = act( sum_p parts[p] + bias )
// ---------------------------------------------------------------------------
__global__ void reducek_k(const float* __restrict__ parts, long long strideP,
                          int nparts, const float* __restrict__ bias,
                          float* __restrict__ outp, int total, int N, int act)
{
    int idx = blockIdx.x * 256 + threadIdx.x;
    if (idx >= total) return;
    float v = 0.f;
    for (int p = 0; p < nparts; ++p) v += parts[(long long)p * strideP + idx];
    if (bias) v += bias[idx % N];
    if (act == 1)      v = 1.f / (1.f + __expf(-v));
    else if (act == 2) v = tanhf(v);
    outp[idx] = v;
}

// ---------------------------------------------------------------------------
// Attention row/col dot products.
// msg layout: [e][bt*15+n][1024], idx = e*900 + bt*15 + n ; one block per idx.
// ---------------------------------------------------------------------------
__global__ __launch_bounds__(256)
void rowcol_k(const float* __restrict__ msg, const float* __restrict__ Watt,
              float* __restrict__ rowo, float* __restrict__ colo)
{
    __shared__ float sR[256], sC[256];
    int idx = blockIdx.x;                // 0 .. 6299
    int e   = idx / 900;
    const float* m  = msg + (long long)idx * 1024;
    const float* wr = Watt + e * 2048;
    const float* wc = wr + 1024;
    float aR = 0.f, aC = 0.f;
    for (int d = threadIdx.x; d < 1024; d += 256) {
        float mv = m[d];
        aR += mv * wr[d];
        aC += mv * wc[d];
    }
    sR[threadIdx.x] = aR; sC[threadIdx.x] = aC;
    __syncthreads();
    for (int s = 128; s > 0; s >>= 1) {
        if (threadIdx.x < s) {
            sR[threadIdx.x] += sR[threadIdx.x + s];
            sC[threadIdx.x] += sC[threadIdx.x + s];
        }
        __syncthreads();
    }
    if (threadIdx.x == 0) { rowo[idx] = sR[0]; colo[idx] = sC[0]; }
}

// scores[bt][e][i][j] = sigmoid(row[e,bt,i] + col[e,bt,j] + b_att[e])
__global__ void scores_k(const float* __restrict__ rowo,
                         const float* __restrict__ colo,
                         const float* __restrict__ batt,
                         float* __restrict__ sc)
{
    int idx = blockIdx.x * blockDim.x + threadIdx.x;
    if (idx >= 60 * 7 * 15 * 15) return;
    int j  = idx % 15;
    int i  = (idx / 15) % 15;
    int e  = (idx / 225) % 7;
    int bt = idx / 1575;
    float x = rowo[e * 900 + bt * 15 + i] + colo[e * 900 + bt * 15 + j] + batt[e];
    sc[idx] = 1.f / (1.f + __expf(-x));
}

// merged[bt*15+i][d] = sum_e sum_j scores[bt][e][i][j] * msg[e][bt*15+j][d]
__global__ __launch_bounds__(256)
void merged_k(const float* __restrict__ sc, const float* __restrict__ msg,
              float* __restrict__ merged)
{
    __shared__ float s[105];
    int bt = blockIdx.x / 15, i = blockIdx.x % 15;
    if (threadIdx.x < 105) {
        int e = threadIdx.x / 15, j = threadIdx.x % 15;
        s[threadIdx.x] = sc[((bt * 7 + e) * 15 + i) * 15 + j];
    }
    __syncthreads();
    for (int d = threadIdx.x; d < 1024; d += 256) {
        float acc = 0.f;
        #pragma unroll
        for (int e = 0; e < 7; ++e)
            #pragma unroll
            for (int j = 0; j < 15; ++j)
                acc += s[e * 15 + j] * msg[(long long)(e * 900 + bt * 15 + j) * 1024 + d];
        merged[(long long)(bt * 15 + i) * 1024 + d] = acc;
    }
}

// abuf[m][0:1024] = merged, abuf[m][1024:2048] = prop
__global__ void concat_k(const float* __restrict__ merged,
                         const float* __restrict__ prop,
                         float* __restrict__ outp, int n)
{
    int idx = blockIdx.x * 256 + threadIdx.x;
    if (idx >= n) return;
    int m = idx >> 11, c = idx & 2047;
    outp[idx] = (c < 1024) ? merged[(m << 10) + c] : prop[(m << 10) + c - 1024];
}

// a2[m][0:1024] = merged, a2[m][1024:2048] = r * prop
__global__ void concat2_k(const float* __restrict__ merged,
                          const float* __restrict__ r,
                          const float* __restrict__ prop,
                          float* __restrict__ outp, int n)
{
    int idx = blockIdx.x * 256 + threadIdx.x;
    if (idx >= n) return;
    int m = idx >> 11, c = idx & 2047;
    outp[idx] = (c < 1024) ? merged[(m << 10) + c]
                           : r[(m << 10) + c - 1024] * prop[(m << 10) + c - 1024];
}

// prop = (1-z)*prop + z*h_hat
__global__ void update_k(const float* __restrict__ z, const float* __restrict__ h,
                         float* __restrict__ prop, int n)
{
    int idx = blockIdx.x * 256 + threadIdx.x;
    if (idx >= n) return;
    float zv = z[idx];
    prop[idx] = (1.f - zv) * prop[idx] + zv * h[idx];
}

// Final outputs from last-round scores. One block per bt (60 blocks).
// out: action (4,15,15,7)=6300 | acty (4,15,6)=360 | actor (4,15,15,15)=13500
__global__ __launch_bounds__(256)
void final_k(const float* __restrict__ sc, float* __restrict__ out)
{
    __shared__ float s[7 * 225];
    __shared__ float sAct[105];
    __shared__ float sW[15];
    __shared__ float sG[7];
    int bt = blockIdx.x;
    for (int l = threadIdx.x; l < 1575; l += 256) s[l] = sc[bt * 1575 + l];
    __syncthreads();
    // actor_score[bt][i][j] = 1 - scores[bt][0][i][j]
    for (int l = threadIdx.x; l < 225; l += 256)
        out[6660 + bt * 225 + l] = 1.f - s[l];
    // action_score[bt][i][e] = sum_j (1 - s[0,i,j]) * s[e,i,j]
    if (threadIdx.x < 105) {
        int i = threadIdx.x / 7, e = threadIdx.x % 7;
        float a = 0.f;
        for (int j = 0; j < 15; ++j)
            a += (1.f - s[i * 15 + j]) * s[e * 225 + i * 15 + j];
        sAct[threadIdx.x] = a;
        out[bt * 105 + threadIdx.x] = a;
    }
    if (threadIdx.x < 15) {
        float w = 0.f;
        for (int j = 0; j < 15; ++j) w += 1.f - s[threadIdx.x * 15 + j];
        sW[threadIdx.x] = w;
    }
    __syncthreads();
    if (threadIdx.x < 7) {
        float g = 0.f;
        for (int i = 0; i < 15; ++i) g += sAct[i * 7 + threadIdx.x] * sW[i];
        sG[threadIdx.x] = g;
    }
    __syncthreads();
    if (threadIdx.x == 0) {
        float mx = -1e30f;
        for (int k = 1; k < 7; ++k) mx = fmaxf(mx, sG[k]);
        float ex[6], sum = 0.f;
        for (int k = 0; k < 6; ++k) { ex[k] = __expf(sG[k + 1] - mx); sum += ex[k]; }
        for (int k = 0; k < 6; ++k) out[6300 + bt * 6 + k] = ex[k] / sum;
    }
}

// ---------------------------------------------------------------------------
extern "C" void kernel_launch(void* const* d_in, const int* in_sizes, int n_in,
                              void* d_out, int out_size, void* d_ws, size_t ws_size,
                              hipStream_t stream)
{
    (void)in_sizes; (void)n_in; (void)out_size; (void)ws_size;
    const float* pose = (const float*)d_in[0];   // (4,15,15,25600)
    const float* Wc   = (const float*)d_in[1];   // (25600,1024)
    const float* bc   = (const float*)d_in[2];   // (1024,)
    const float* We   = (const float*)d_in[3];   // (7,1024,1024)
    const float* be   = (const float*)d_in[4];   // (7,1024)
    const float* Watt = (const float*)d_in[5];   // (7,2048)
    const float* batt = (const float*)d_in[6];   // (7,)
    const float* Wr   = (const float*)d_in[7];   // (2048,1024)
    const float* br   = (const float*)d_in[8];
    const float* Wz   = (const float*)d_in[9];
    const float* bz   = (const float*)d_in[10];
    const float* Wh   = (const float*)d_in[11];
    const float* bh   = (const float*)d_in[12];
    float* out = (float*)d_out;

    const int M = 900, S = 1024, DF = 25600, E = 7;
    const long long MS = (long long)M * S;       // 921600

    float* ws     = (float*)d_ws;
    float* prop   = ws;             ws += MS;                // 900x1024
    float* msg    = ws;             ws += (long long)E * MS; // 7x900x1024
    float* pbuf   = ws;             ws += 4 * MS;            // split-K partials
    float* rowb   = ws;             ws += 6300;
    float* colb   = ws;             ws += 6300;
    float* scb    = ws;             ws += 94500;
    float* merged = ws;             ws += MS;
    float* abuf   = ws;             ws += 2 * MS;            // 900x2048
    float* a2buf  = ws;             ws += 2 * MS;
    float* rbuf   = ws;             ws += MS;
    float* zbuf   = ws;             ws += MS;
    float* hbuf   = ws;             ws += MS;

    dim3 blk(256);
    dim3 gGemmK4(S / TILE_N, (M + TILE_M - 1) / TILE_M, 4);  // 8 x 8 x 4
    dim3 gGemmK2(S / TILE_N, (M + TILE_M - 1) / TILE_M, 2);  // 8 x 8 x 2
    dim3 gEdge(S / TILE_N, (M + TILE_M - 1) / TILE_M, E);    // 8 x 8 x 7
    dim3 gRed(3600);                                         // MS / 256

    // prop = pose @ W_compress + b_compress  (900 x 25600 x 1024, split-K x4)
    gemm_bf16_wmma<<<gGemmK4, blk, 0, stream>>>(pose, DF, DF / 4, Wc, S, 0,
                                                nullptr, 0, pbuf, S, MS,
                                                M, S, DF / 4, 0);
    reducek_k<<<gRed, blk, 0, stream>>>(pbuf, MS, 4, bc, prop, (int)MS, S, 0);

    for (int it = 0; it < 6; ++it) {
        // msg[e] = prop @ W_edge[e] + b_edge[e]  (7 batched 900x1024x1024)
        gemm_bf16_wmma<<<gEdge, blk, 0, stream>>>(prop, S, 0, We, S,
                                                  (long long)S * S, be, S,
                                                  msg, S, MS, M, S, S, 0);
        rowcol_k<<<dim3(6300), blk, 0, stream>>>(msg, Watt, rowb, colb);
        scores_k<<<dim3((94500 + 255) / 256), blk, 0, stream>>>(rowb, colb, batt, scb);

        if (it < 5) {  // last round's GRU state is never consumed
            merged_k<<<dim3(M), blk, 0, stream>>>(scb, msg, merged);
            concat_k<<<dim3(7200), blk, 0, stream>>>(merged, prop, abuf, 2 * (int)MS);
            // r / z gates: 900 x 2048 x 1024, split-K x2
            gemm_bf16_wmma<<<gGemmK2, blk, 0, stream>>>(abuf, 2 * S, S, Wr, S, 0,
                                                        nullptr, 0, pbuf, S, MS,
                                                        M, S, S, 0);
            reducek_k<<<gRed, blk, 0, stream>>>(pbuf, MS, 2, br, rbuf, (int)MS, S, 1);
            gemm_bf16_wmma<<<gGemmK2, blk, 0, stream>>>(abuf, 2 * S, S, Wz, S, 0,
                                                        nullptr, 0, pbuf, S, MS,
                                                        M, S, S, 0);
            reducek_k<<<gRed, blk, 0, stream>>>(pbuf, MS, 2, bz, zbuf, (int)MS, S, 1);
            concat2_k<<<dim3(7200), blk, 0, stream>>>(merged, rbuf, prop, a2buf, 2 * (int)MS);
            gemm_bf16_wmma<<<gGemmK2, blk, 0, stream>>>(a2buf, 2 * S, S, Wh, S, 0,
                                                        nullptr, 0, pbuf, S, MS,
                                                        M, S, S, 0);
            reducek_k<<<gRed, blk, 0, stream>>>(pbuf, MS, 2, bh, hbuf, (int)MS, S, 2);
            update_k<<<dim3(3600), blk, 0, stream>>>(zbuf, hbuf, prop, (int)MS);
        }
    }

    final_k<<<dim3(60), blk, 0, stream>>>(scb, out);
}
